// GraphSAGE_PPI_62663572848802
// MI455X (gfx1250) — matrix operand order repaired and checked by reading, not compiled, not run on hardware
//
#include <hip/hip_runtime.h>
#include <math.h>

#define NN   50000
#define NE   800000
#define IND  50
#define HID  256
#define NC   121
#define NCP  128   // classes padded to 8 n-tiles
#define KP1  64    // in-dim padded to 2 k-steps

// ---------- types for WMMA ----------
typedef __attribute__((ext_vector_type(16))) __bf16 bf16x16;
typedef __attribute__((ext_vector_type(8)))  float  f32x8;
typedef __attribute__((ext_vector_type(8)))  unsigned short u16x8;

union Frag { bf16x16 v; u16x8 h[2]; };

__device__ __forceinline__ unsigned short f2bf(float f) {
  unsigned int u = __float_as_uint(f);
  u += 0x7FFFu + ((u >> 16) & 1u);          // round-to-nearest-even
  return (unsigned short)(u >> 16);
}
__device__ __forceinline__ float bf2f(unsigned short h) {
  return __uint_as_float(((unsigned int)h) << 16);
}

// A fragment: 16x32 bf16, row-major source, stride KP.
// lane 0-15: row=lane, K=kb..kb+7 & kb+16..kb+23 with kb = (lane>>4)*8
__device__ __forceinline__ Frag load_a(const unsigned short* __restrict__ A,
                                       int KP, int m0, int k0, int lane) {
  int row = lane & 15;
  int kb  = (lane >> 4) << 3;
  const unsigned short* p = A + (size_t)(m0 + row) * KP + (size_t)(k0 + kb);
  Frag f;
  f.h[0] = *reinterpret_cast<const u16x8*>(p);
  f.h[1] = *reinterpret_cast<const u16x8*>(p + 16);
  return f;
}

// B fragment: 32x16 bf16 from TRANSPOSED weights WT[N][K] (k contiguous).
// lane col = lane&15, K=kb..kb+7 & kb+16..kb+23 with kb = (lane>>4)*8.
__device__ __forceinline__ Frag load_b(const unsigned short* __restrict__ WT,
                                       int KP, int n0, int k0, int lane) {
  int col = lane & 15;
  int kb  = (lane >> 4) << 3;
  const unsigned short* p = WT + (size_t)(n0 + col) * KP + (size_t)(k0 + kb);
  Frag f;
  f.h[0] = *reinterpret_cast<const u16x8*>(p);
  f.h[1] = *reinterpret_cast<const u16x8*>(p + 16);
  return f;
}

__device__ __forceinline__ f32x8 wmma_bf16(const Frag& a, const Frag& b, f32x8 c) {
  return __builtin_amdgcn_wmma_f32_16x16x32_bf16(false, a.v, false, b.v,
                                                 (short)0, c, false, false);
}

// ---------- scatter / prep kernels ----------
__global__ void k_deg(const int* __restrict__ dst, float* __restrict__ deg) {
  int e = blockIdx.x * 256 + threadIdx.x;
  if (e < NE) atomicAdd(&deg[dst[e]], 1.0f);
}

__global__ void k_inv(float* __restrict__ deg) {
  int i = blockIdx.x * 256 + threadIdx.x;
  if (i < NN) deg[i] = 1.0f / fmaxf(deg[i], 1.0f);
}

// one thread per (edge, col<64); 50 cols active
__global__ void k_scat1(const int* __restrict__ src, const int* __restrict__ dst,
                        const float* __restrict__ x, float* __restrict__ agg1) {
  int t = blockIdx.x * 256 + threadIdx.x;
  int e = t >> 6, c = t & 63;
  if (c < IND) {
    int s = src[e], d = dst[e];
    atomicAdd(&agg1[(size_t)d * KP1 + c], x[(size_t)s * IND + c]);
  }
}

// Xb (padded bf16 x) and Mb (padded bf16 mean1); t over NN*64
__global__ void k_prep1(const float* __restrict__ x, const float* __restrict__ agg1,
                        const float* __restrict__ invdeg,
                        unsigned short* __restrict__ Xb, unsigned short* __restrict__ Mb) {
  int t = blockIdx.x * 256 + threadIdx.x;
  int i = t >> 6, c = t & 63;
  float xv = (c < IND) ? x[(size_t)i * IND + c] : 0.0f;
  Xb[t] = f2bf(xv);
  Mb[t] = f2bf(agg1[t] * invdeg[i]);   // agg1 pad cols are memset-0
}

// layer-1 weights (50x256) -> transposed padded bf16 [3][256][64]; t == flat dst index
__global__ void k_w1(const float* __restrict__ w1l, const float* __restrict__ w1r,
                     const float* __restrict__ wl1, unsigned short* __restrict__ W1T) {
  int t = blockIdx.x * 256 + threadIdx.x;       // 3*256*64
  int k = t & 63;
  int na = t >> 6;                              // which*256 + n
  int which = na >> 8, n = na & 255;
  const float* w = (which == 0) ? w1l : ((which == 1) ? w1r : wl1);
  float v = (k < IND) ? w[(size_t)k * HID + n] : 0.0f;
  W1T[t] = f2bf(v);
}

// layer-2 weights (256x121) -> transposed padded bf16 [3][128][256]
__global__ void k_w2(const float* __restrict__ w2l, const float* __restrict__ w2r,
                     const float* __restrict__ wl2, unsigned short* __restrict__ W2T) {
  int t = blockIdx.x * 256 + threadIdx.x;       // 3*128*256
  int k = t & 255;
  int na = t >> 8;                              // which*128 + n
  int which = na >> 7, n = na & 127;
  const float* w = (which == 0) ? w2l : ((which == 1) ? w2r : wl2);
  float v = (n < NC) ? w[(size_t)k * NC + n] : 0.0f;
  W2T[t] = f2bf(v);
}

__global__ void k_bias2(const float* __restrict__ b2, const float* __restrict__ bl2,
                        float* __restrict__ b2p, float* __restrict__ bl2p) {
  int t = threadIdx.x;  // 128
  b2p[t]  = (t < NC) ? b2[t]  : 0.0f;
  bl2p[t] = (t < NC) ? bl2[t] : 0.0f;
}

// ---------- fused GEMM layer 1: sage = Mb@W1l + Xb@W1r + b1 ; lin = Xb@Wl1 + bl1
__global__ void k_gemm1(const unsigned short* __restrict__ Mb, const unsigned short* __restrict__ Xb,
                        const unsigned short* __restrict__ W1T,
                        const float* __restrict__ b1, const float* __restrict__ bl1,
                        float* __restrict__ S1, float* __restrict__ L1) {
  int wave = (blockIdx.x * 256 + threadIdx.x) >> 5;
  int lane = threadIdx.x & 31;
  int mt = wave >> 4, nt = wave & 15;           // 16 n-tiles
  int m0 = mt * 16, n0 = nt * 16;
  int col = lane & 15;
  const unsigned short* W1lT = W1T;
  const unsigned short* W1rT = W1T + 256 * 64;
  const unsigned short* Wl1T = W1T + 2 * 256 * 64;
  float bs = b1[n0 + col], bl = bl1[n0 + col];
  f32x8 cs, cl;
#pragma unroll
  for (int r = 0; r < 8; ++r) { cs[r] = bs; cl[r] = bl; }
#pragma unroll
  for (int k0 = 0; k0 < KP1; k0 += 32) {
    Frag am = load_a(Mb, KP1, m0, k0, lane);
    Frag ax = load_a(Xb, KP1, m0, k0, lane);
    Frag bwl = load_b(W1lT, KP1, n0, k0, lane);
    Frag bwr = load_b(W1rT, KP1, n0, k0, lane);
    Frag bln = load_b(Wl1T, KP1, n0, k0, lane);
    cs = wmma_bf16(am, bwl, cs);
    cs = wmma_bf16(ax, bwr, cs);
    cl = wmma_bf16(ax, bln, cl);
  }
  int rb = (lane >> 4) * 8;
#pragma unroll
  for (int r = 0; r < 8; ++r) {
    size_t idx = (size_t)(m0 + rb + r) * HID + (n0 + col);
    S1[idx] = cs[r];
    L1[idx] = cl[r];
  }
}

// epilogue 1: wave per row; h = elu(sage/max(||sage||,eps) + lin) -> bf16
__global__ void k_ep1(const float* __restrict__ S1, const float* __restrict__ L1,
                      unsigned short* __restrict__ Hb) {
  int row  = (blockIdx.x * 256 + threadIdx.x) >> 5;
  int lane = threadIdx.x & 31;
  const float* s = S1 + (size_t)row * HID;
  const float* l = L1 + (size_t)row * HID;
  float v[8]; float ss = 0.0f;
#pragma unroll
  for (int j = 0; j < 8; ++j) { float t = s[lane * 8 + j]; v[j] = t; ss += t * t; }
#pragma unroll
  for (int m = 16; m >= 1; m >>= 1) ss += __shfl_xor(ss, m, 32);
  float inv = 1.0f / fmaxf(sqrtf(ss), 1e-12f);
  unsigned short* h = Hb + (size_t)row * HID;
#pragma unroll
  for (int j = 0; j < 8; ++j) {
    float t = v[j] * inv + l[lane * 8 + j];
    t = (t > 0.0f) ? t : (expf(t) - 1.0f);    // ELU(alpha=1)
    h[lane * 8 + j] = f2bf(t);
  }
}

// layer-2 scatter: one thread per (edge, 4 cols of 256)
__global__ void k_scat2(const int* __restrict__ src, const int* __restrict__ dst,
                        const unsigned short* __restrict__ Hb, float* __restrict__ agg2) {
  int t = blockIdx.x * 256 + threadIdx.x;
  int e = t >> 6, c = (t & 63) << 2;
  int s = src[e], d = dst[e];
  const unsigned short* hp = Hb + (size_t)s * HID + c;
  float* ap = agg2 + (size_t)d * HID + c;
#pragma unroll
  for (int j = 0; j < 4; ++j) atomicAdd(&ap[j], bf2f(hp[j]));
}

// mean2 -> bf16; t over NN*256
__global__ void k_prep2(const float* __restrict__ agg2, const float* __restrict__ invdeg,
                        unsigned short* __restrict__ M2b) {
  int t = blockIdx.x * 256 + threadIdx.x;
  int i = t >> 8;
  M2b[t] = f2bf(agg2[t] * invdeg[i]);
}

// ---------- fused GEMM layer 2 (K=256, N padded to 128)
__global__ void k_gemm2(const unsigned short* __restrict__ M2b, const unsigned short* __restrict__ Hb,
                        const unsigned short* __restrict__ W2T,
                        const float* __restrict__ b2p, const float* __restrict__ bl2p,
                        float* __restrict__ S2, float* __restrict__ L2o) {
  int wave = (blockIdx.x * 256 + threadIdx.x) >> 5;
  int lane = threadIdx.x & 31;
  int mt = wave >> 3, nt = wave & 7;            // 8 n-tiles
  int m0 = mt * 16, n0 = nt * 16;
  int col = lane & 15;
  const unsigned short* W2lT = W2T;
  const unsigned short* W2rT = W2T + 128 * 256;
  const unsigned short* Wl2T = W2T + 2 * 128 * 256;
  float bs = b2p[n0 + col], bl = bl2p[n0 + col];
  f32x8 cs, cl;
#pragma unroll
  for (int r = 0; r < 8; ++r) { cs[r] = bs; cl[r] = bl; }
#pragma unroll
  for (int k0 = 0; k0 < HID; k0 += 32) {
    Frag am = load_a(M2b, HID, m0, k0, lane);
    Frag ah = load_a(Hb,  HID, m0, k0, lane);
    Frag bwl = load_b(W2lT, HID, n0, k0, lane);
    Frag bwr = load_b(W2rT, HID, n0, k0, lane);
    Frag bln = load_b(Wl2T, HID, n0, k0, lane);
    cs = wmma_bf16(am, bwl, cs);
    cs = wmma_bf16(ah, bwr, cs);
    cl = wmma_bf16(ah, bln, cl);
  }
  int rb = (lane >> 4) * 8;
#pragma unroll
  for (int r = 0; r < 8; ++r) {
    size_t idx = (size_t)(m0 + rb + r) * NCP + (n0 + col);
    S2[idx] = cs[r];
    L2o[idx] = cl[r];
  }
}

// epilogue 2: wave per row; out = sage2/max(||sage2||,eps) + lin2  (cols < 121)
__global__ void k_ep2(const float* __restrict__ S2, const float* __restrict__ L2o,
                      float* __restrict__ out) {
  int row  = (blockIdx.x * 256 + threadIdx.x) >> 5;
  int lane = threadIdx.x & 31;
  const float* s = S2  + (size_t)row * NCP;
  const float* l = L2o + (size_t)row * NCP;
  float v[4]; float ss = 0.0f;
#pragma unroll
  for (int j = 0; j < 4; ++j) { float t = s[lane * 4 + j]; v[j] = t; ss += t * t; }
#pragma unroll
  for (int m = 16; m >= 1; m >>= 1) ss += __shfl_xor(ss, m, 32);
  float inv = 1.0f / fmaxf(sqrtf(ss), 1e-12f);
#pragma unroll
  for (int j = 0; j < 4; ++j) {
    int c = lane * 4 + j;
    if (c < NC) out[(size_t)row * NC + c] = v[j] * inv + l[lane * 4 + j];
  }
}

// ---------- workspace layout ----------
static constexpr size_t A256(size_t x) { return (x + 255) & ~size_t(255); }
static constexpr size_t OFF_DEG  = 0;
static constexpr size_t OFF_W1T  = A256(OFF_DEG  + (size_t)NN * 4);
static constexpr size_t OFF_W2T  = A256(OFF_W1T  + (size_t)3 * 256 * 64 * 2);
static constexpr size_t OFF_B2P  = A256(OFF_W2T  + (size_t)3 * 128 * 256 * 2);
static constexpr size_t OFF_BL2P = A256(OFF_B2P  + 128 * 4);
static constexpr size_t OFF_XB   = A256(OFF_BL2P + 128 * 4);
static constexpr size_t OFF_MB   = A256(OFF_XB   + (size_t)NN * KP1 * 2);
static constexpr size_t OFF_AGG1 = A256(OFF_MB   + (size_t)NN * KP1 * 2);
static constexpr size_t OFF_HB   = A256(OFF_AGG1 + (size_t)NN * KP1 * 4);
static constexpr size_t OFF_M2B  = A256(OFF_HB   + (size_t)NN * HID * 2);
static constexpr size_t OFF_BIG1 = A256(OFF_M2B  + (size_t)NN * HID * 2); // S1, then agg2
static constexpr size_t OFF_BIG2 = A256(OFF_BIG1 + (size_t)NN * HID * 4); // L1, then S2+L2o

extern "C" void kernel_launch(void* const* d_in, const int* in_sizes, int n_in,
                              void* d_out, int out_size, void* d_ws, size_t ws_size,
                              hipStream_t stream) {
  const float* x    = (const float*)d_in[0];
  const int*   ei   = (const int*)d_in[1];
  const float* w1l  = (const float*)d_in[2];
  const float* b1   = (const float*)d_in[3];
  const float* w1r  = (const float*)d_in[4];
  const float* wl1  = (const float*)d_in[5];
  const float* bl1  = (const float*)d_in[6];
  const float* w2l  = (const float*)d_in[7];
  const float* b2   = (const float*)d_in[8];
  const float* w2r  = (const float*)d_in[9];
  const float* wl2  = (const float*)d_in[10];
  const float* bl2  = (const float*)d_in[11];
  const int* src = ei;
  const int* dst = ei + NE;

  char* ws = (char*)d_ws;
  float*          deg  = (float*)(ws + OFF_DEG);
  unsigned short* W1T  = (unsigned short*)(ws + OFF_W1T);
  unsigned short* W2T  = (unsigned short*)(ws + OFF_W2T);
  float*          b2p  = (float*)(ws + OFF_B2P);
  float*          bl2p = (float*)(ws + OFF_BL2P);
  unsigned short* Xb   = (unsigned short*)(ws + OFF_XB);
  unsigned short* Mb   = (unsigned short*)(ws + OFF_MB);
  float*          agg1 = (float*)(ws + OFF_AGG1);
  unsigned short* Hb   = (unsigned short*)(ws + OFF_HB);
  unsigned short* M2b  = (unsigned short*)(ws + OFF_M2B);
  float*          S1   = (float*)(ws + OFF_BIG1);            // layer-1 sage f32
  float*          agg2 = (float*)(ws + OFF_BIG1);            // reuse after ep1
  float*          L1   = (float*)(ws + OFF_BIG2);            // layer-1 lin f32
  float*          S2   = (float*)(ws + OFF_BIG2);            // reuse after ep1
  float*          L2o  = (float*)(ws + OFF_BIG2 + (size_t)NN * NCP * 4);
  float*          out  = (float*)d_out;

  hipMemsetAsync(deg,  0, (size_t)NN * 4, stream);
  hipMemsetAsync(agg1, 0, (size_t)NN * KP1 * 4, stream);

  k_deg  <<<NE / 256, 256, 0, stream>>>(dst, deg);
  k_scat1<<<(NE * 64) / 256, 256, 0, stream>>>(src, dst, x, agg1);
  k_inv  <<<(NN + 255) / 256, 256, 0, stream>>>(deg);
  k_prep1<<<(NN * 64) / 256, 256, 0, stream>>>(x, agg1, deg, Xb, Mb);
  k_w1   <<<(3 * 256 * 64) / 256, 256, 0, stream>>>(w1l, w1r, wl1, W1T);
  k_w2   <<<(3 * 128 * 256) / 256, 256, 0, stream>>>(w2l, w2r, wl2, W2T);
  k_bias2<<<1, 128, 0, stream>>>(b2, bl2, b2p, bl2p);

  k_gemm1<<<(NN / 16) * 16 / 8, 256, 0, stream>>>(Mb, Xb, W1T, b1, bl1, S1, L1);
  k_ep1  <<<NN / 8, 256, 0, stream>>>(S1, L1, Hb);

  hipMemsetAsync(agg2, 0, (size_t)NN * HID * 4, stream);
  k_scat2<<<(NE * 64) / 256, 256, 0, stream>>>(src, dst, Hb, agg2);
  k_prep2<<<(NN * HID) / 256, 256, 0, stream>>>(agg2, deg, M2b);

  k_gemm2<<<(NN / 16) * 8 / 8, 256, 0, stream>>>(M2b, Hb, W2T, b2p, bl2p, S2, L2o);
  k_ep2  <<<NN / 8, 256, 0, stream>>>(S2, L2o, out);
}
